// FFF_46316927320395
// MI455X (gfx1250) — compile-verified, matrix operand order checked by reading, not run on hardware
//
#include <hip/hip_runtime.h>
#include <cstdint>

// ---------------- problem constants ----------------
#define TOK_PER_WG 16
#define D          2048      // D_IN == D_OUT
#define NLVL       12        // DEPTH+1
#define NNODES     4095

// LDS tile strides (in floats). xs: +4 pad -> 4-bank row skew (conflict-free
// WMMA A loads). ws: TDM pad (1 dword per 256) -> stride 2056, 8-bank skew.
#define XS_STRIDE  2052
#define WS_STRIDE  2056

typedef __attribute__((ext_vector_type(2)))  float        v2f;
typedef __attribute__((ext_vector_type(8)))  float        v8f;
typedef __attribute__((ext_vector_type(16))) _Float16     v16h;
typedef __attribute__((ext_vector_type(4)))  unsigned int v4u;
typedef __attribute__((ext_vector_type(4)))  int          v4i;
typedef __attribute__((ext_vector_type(8)))  int          v8i;

// ---------------- dynamic LDS layout (bytes) ----------------
#define XS_BYTES   (TOK_PER_WG * XS_STRIDE * 4)   // 131328
#define WS_BYTES   (TOK_PER_WG * WS_STRIDE * 4)   // 131584
#define OFF_XS     0
#define OFF_WS     (OFF_XS + XS_BYTES)
#define OFF_PART   (OFF_WS + WS_BYTES)            // 8 waves x 16 partial diagonals
#define OFF_CUR    (OFF_PART + 8 * 16 * 4)
#define OFF_GEL    (OFF_CUR + TOK_PER_WG * 4)
#define OFF_NS     (OFF_GEL + TOK_PER_WG * NLVL * 4)
#define SMEM_BYTES (OFF_NS + TOK_PER_WG * NLVL * 4)   // ~265 KB (<320 KB/WGP)

__device__ __forceinline__ float gelu_exact(float v) {
    return 0.5f * v * (1.0f + erff(v * 0.70710678118654752440f));
}

// ---------------- w_out transpose: wT[n][d] = w_out[d][n] ----------------
__global__ __launch_bounds__(256) void wout_transpose(const float* __restrict__ w_out,
                                                      float* __restrict__ wT) {
    __shared__ float tile[32][33];
    const int n0 = blockIdx.x * 32;
    const int d0 = blockIdx.y * 32;
    const int tx = threadIdx.x & 31;
    const int ty = threadIdx.x >> 5;            // 32x8 threads
    for (int i = ty; i < 32; i += 8) {
        int d = d0 + i, n = n0 + tx;
        tile[i][tx] = (n < NNODES) ? w_out[(size_t)d * NNODES + n] : 0.0f;
    }
    __syncthreads();
    for (int i = ty; i < 32; i += 8) {
        int n = n0 + i, d = d0 + tx;
        if (n < NNODES) wT[(size_t)n * D + d] = tile[tx][i];
    }
}

// ---------------- main fused FFF kernel: 16 tokens / workgroup ----------------
__global__ __launch_bounds__(256) void fff_main(const float* __restrict__ x,
                                                const float* __restrict__ w_in,
                                                const float* __restrict__ w_out,
                                                const float* __restrict__ wT,
                                                float* __restrict__ out,
                                                int useT) {
    extern __shared__ __align__(16) char smem[];
    float* xs   = (float*)(smem + OFF_XS);
    float* wsb  = (float*)(smem + OFF_WS);
    float* part = (float*)(smem + OFF_PART);
    int*   cur  = (int*)  (smem + OFF_CUR);
    float* gel  = (float*)(smem + OFF_GEL);
    int*   ns   = (int*)  (smem + OFF_NS);

    const int tid  = threadIdx.x;
    const int tok0 = blockIdx.x * TOK_PER_WG;

    // ---- phase 0: stage x tile (16 x 2048 f32) into padded LDS, read x ONCE ----
    for (int i = tid; i < TOK_PER_WG * (D / 4); i += 256) {
        int r = i >> 9, c4 = i & 511;
        float4 v = ((const float4*)(x + (size_t)(tok0 + r) * D))[c4];
        *((float4*)(xs + r * XS_STRIDE + c4 * 4)) = v;
    }
    if (tid < TOK_PER_WG) cur[tid] = 0;
    __syncthreads();

    const int lane = tid & 31;
    const int wv   = tid >> 5;       // wave id 0..7, each owns K-chunk of 256
    const int row  = lane & 15;      // token (M for A, N for B)
    const int hf   = lane >> 4;      // fragment half select
    const float* xrow  = xs  + row * XS_STRIDE;
    const float* wrow  = wsb + row * WS_STRIDE;
    const int    kbase = wv * 256;

    // ---- phase 1: 12 tree levels ----
    for (int lvl = 0; lvl < NLVL; ++lvl) {
        // --- gather 16 data-dependent w_in rows into LDS ---
#if __has_builtin(__builtin_amdgcn_tensor_load_to_lds)
        if (tid < 32) {
            // TDM gather-mode descriptor (D#), 16-bit row indices = tree nodes.
            unsigned ldsaddr = (unsigned)(uintptr_t)wsb;   // flat addr low 32b == LDS offset
            unsigned long long ga = (unsigned long long)(uintptr_t)w_in;
            v4u g0;
            g0[0] = 0x80000001u;                           // count=1, gather_mode=1, 16b idx
            g0[1] = ldsaddr;                               // lds_addr
            g0[2] = (unsigned)ga;                          // global_addr[31:0]
            g0[3] = ((unsigned)(ga >> 32) & 0x01FFFFFFu) | 0x80000000u; // addr[56:32], type=2
            v8i g1;
            g1[0] = (2 << 16) | (1 << 20) | (7 << 22);     // data_size=4B, pad_en, intvl=256, amt=1
            g1[1] = (int)(2048u << 16);                    // tensor_dim0 = 2048 (bits 79:48)
            g1[2] = (int)(4095u << 16);                    // tensor_dim1 = 4095 (bits 111:80)
            g1[3] = (int)(2048u << 16);                    // tile_dim0 = 2048 (bits 127:112)
            g1[4] = 16;                                    // tile_dim1 = #indices
            g1[5] = 2048;                                  // tensor_dim0_stride
            g1[6] = 0;  g1[7] = 0;
            v4i g2, g3;
#pragma unroll
            for (int j = 0; j < 4; ++j) {
                g2[j] = (cur[2 * j]     & 0xFFFF) | (cur[2 * j + 1] << 16);
                g3[j] = (cur[8 + 2 * j] & 0xFFFF) | (cur[8 + 2 * j + 1] << 16);
            }
#if defined(__clang_major__) && (__clang_major__ >= 23)
            v8i g4 = {0, 0, 0, 0, 0, 0, 0, 0};
            __builtin_amdgcn_tensor_load_to_lds(g0, g1, g2, g3, g4, 0);
#else
            __builtin_amdgcn_tensor_load_to_lds(g0, g1, g2, g3, 0);
#endif
        }
#if __has_builtin(__builtin_amdgcn_s_wait_tensorcnt)
        __builtin_amdgcn_s_wait_tensorcnt(0);
#else
        asm volatile("s_wait_tensorcnt 0x0" ::: "memory");
#endif
#else   // no TDM builtin: coalesced manual gather into the same padded layout
        for (int i = tid; i < TOK_PER_WG * (D / 4); i += 256) {
            int r = i >> 9, c4i = (i & 511) * 4;
            const float* src = w_in + (size_t)cur[r] * D + c4i;
            float* dst = wsb + r * WS_STRIDE + c4i + (c4i >> 8);
            dst[0] = src[0]; dst[1] = src[1]; dst[2] = src[2]; dst[3] = src[3];
        }
#endif
        __syncthreads();

        // --- 16 scores = diagonal of X_tile(16xK) * W_tile^T(Kx16) via WMMA ---
        v8f c = {};
#if __has_builtin(__builtin_amdgcn_wmma_f32_16x16x4_f32)
#pragma unroll 8
        for (int s = 0; s < 64; ++s) {
            int k0 = kbase + s * 4;
            float2 av = *(const float2*)(xrow + k0 + 2 * hf);
            v2f a; a[0] = av.x; a[1] = av.y;
            int wk = k0 + (k0 >> 8) + 2 * hf;               // ws pad: +1 dword / 256
            v2f b; b[0] = wrow[wk]; b[1] = wrow[wk + 1];
            c = __builtin_amdgcn_wmma_f32_16x16x4_f32(false, a, false, b,
                                                      (short)0, c, false, false);
        }
#else   // fallback: codegen-confirmed f16 WMMA (reduced precision)
        for (int s = 0; s < 8; ++s) {
            int k0 = kbase + s * 32;
            v16h a, b;
#pragma unroll
            for (int i = 0; i < 8; ++i) {
                int c1 = k0 + hf * 8 + i;
                int c2 = k0 + 16 + hf * 8 + i;
                a[i]     = (_Float16)xrow[c1];
                a[8 + i] = (_Float16)xrow[c2];
                b[i]     = (_Float16)wrow[c1 + (c1 >> 8)];
                b[8 + i] = (_Float16)wrow[c2 + (c2 >> 8)];
            }
            c = __builtin_amdgcn_wmma_f32_16x16x32_f16(false, a, false, b,
                                                       (short)0, c, false, false);
        }
#endif
        // --- extract diagonal (t,t): t<8 -> lane t, VGPR t; t>=8 -> lane t+16, VGPR t-8 ---
        if (lane < 8 || lane >= 24) {
            int k = lane & 7;
            float dv;
            switch (k) {
                case 0: dv = c[0]; break; case 1: dv = c[1]; break;
                case 2: dv = c[2]; break; case 3: dv = c[3]; break;
                case 4: dv = c[4]; break; case 5: dv = c[5]; break;
                case 6: dv = c[6]; break; default: dv = c[7]; break;
            }
            int slot = (lane < 8) ? lane : (lane - 16);
            part[wv * 16 + slot] = dv;                      // fixed writer per (wave,slot)
        }
        __syncthreads();

        // --- deterministic fixed-order reduction + tree step ---
        if (tid < TOK_PER_WG) {
            float s = 0.0f;
#pragma unroll
            for (int w = 0; w < 8; ++w) s += part[w * 16 + tid];
            ns [tid * NLVL + lvl] = cur[tid];
            gel[tid * NLVL + lvl] = gelu_exact(s);
            cur[tid] = cur[tid] * 2 + (s >= 0.0f ? 1 : 0) + 1;
        }
        __syncthreads();
    }

    // ---- phase 2: out[t][:] = sum_l gelu_l * w_out[:, node_l] (L2-resident gathers) ----
    for (int t = 0; t < TOK_PER_WG; ++t) {
        float a0 = 0, a1 = 0, a2 = 0, a3 = 0;
        float b0 = 0, b1 = 0, b2 = 0, b3 = 0;
#pragma unroll
        for (int l = 0; l < NLVL; ++l) {
            float g = gel[t * NLVL + l];
            int   n = ns [t * NLVL + l];
            if (useT) {
                const float4* rw = (const float4*)(wT + (size_t)n * D);
                float4 u = rw[tid];
                float4 v = rw[tid + 256];
                a0 += g * u.x; a1 += g * u.y; a2 += g * u.z; a3 += g * u.w;
                b0 += g * v.x; b1 += g * v.y; b2 += g * v.z; b3 += g * v.w;
            } else {
                const float* base = w_out + n;              // strided fallback
                int d0i = tid * 4, d1i = 1024 + tid * 4;
                a0 += g * base[(size_t)(d0i + 0) * NNODES];
                a1 += g * base[(size_t)(d0i + 1) * NNODES];
                a2 += g * base[(size_t)(d0i + 2) * NNODES];
                a3 += g * base[(size_t)(d0i + 3) * NNODES];
                b0 += g * base[(size_t)(d1i + 0) * NNODES];
                b1 += g * base[(size_t)(d1i + 1) * NNODES];
                b2 += g * base[(size_t)(d1i + 2) * NNODES];
                b3 += g * base[(size_t)(d1i + 3) * NNODES];
            }
        }
        float4* o = (float4*)(out + (size_t)(tok0 + t) * D);
        o[tid]       = make_float4(a0, a1, a2, a3);
        o[tid + 256] = make_float4(b0, b1, b2, b3);
    }
}

extern "C" void kernel_launch(void* const* d_in, const int* in_sizes, int n_in,
                              void* d_out, int out_size, void* d_ws, size_t ws_size,
                              hipStream_t stream) {
    (void)n_in; (void)out_size;
    const float* x     = (const float*)d_in[0];
    const float* w_in  = (const float*)d_in[1];
    const float* w_out = (const float*)d_in[2];
    float* out = (float*)d_out;
    float* wT  = (float*)d_ws;

    const int useT = (ws_size >= (size_t)NNODES * D * sizeof(float)) ? 1 : 0;
    if (useT) {
        dim3 grid((NNODES + 31) / 32, D / 32);
        wout_transpose<<<grid, 256, 0, stream>>>(w_out, wT);
    }

    (void)hipFuncSetAttribute(reinterpret_cast<const void*>(fff_main),
                              hipFuncAttributeMaxDynamicSharedMemorySize,
                              (int)SMEM_BYTES);

    const int nTok = in_sizes[0] / D;                   // B*S = 8192
    fff_main<<<nTok / TOK_PER_WG, 256, SMEM_BYTES, stream>>>(x, w_in, w_out, wT, out, useT);
}
